// QIGAT_39608188404043
// MI455X (gfx1250) — matrix operand order repaired and checked by reading, not compiled, not run on hardware
//
#include <hip/hip_runtime.h>
#include <hip/hip_bf16.h>

// ---------------------------------------------------------------------------
// QIGAT forward for MI455X (gfx1250, wave32).
// Dense layers: v_wmma_f32_16x16x32_f16, f16 in / f32 accum.
//   A row-major (contiguous 128-bit frag loads), B pre-transposed to K-major
//   so B frags are also contiguous 128-bit loads. NT C-tiles per wave,
//   software-pipelined: next k-step's fragments load while current WMMAs run.
// Edge phase: 1 block / edge, 1 wave / head, f32 global atomics for scatter.
// ---------------------------------------------------------------------------

typedef __attribute__((ext_vector_type(16))) _Float16     v16h;
typedef __attribute__((ext_vector_type(8)))  float        v8f;
typedef __attribute__((ext_vector_type(4)))  unsigned int v4u;

union frag16 { v16h h; v4u u[2]; };

#define DIN   128
#define QOUT  256
#define TOTQ  288
#define HID   32
#define NHEAD 4
#define HD    128           // NHEAD*HID
#define KPAIR 32
#define LN_EPS 1e-5f
#define PI_F  3.14159265358979f

// ---------------- elementwise helpers ----------------
__device__ __forceinline__ float elu1(float x) { return x > 0.f ? x : expm1f(x); }

// ---------------- f32 -> f16 convert (row-major copy) ----------------
__global__ void qigat_cvt_f16(const float* __restrict__ in, _Float16* __restrict__ out, int n) {
    int i = blockIdx.x * blockDim.x + threadIdx.x;
    int stride = gridDim.x * blockDim.x;
    for (; i < n; i += stride) out[i] = (_Float16)in[i];
}

// ---------------- f32 [R,C] -> f16 transposed [C,R] (weights only) --------
__global__ void qigat_cvt_f16_t(const float* __restrict__ in, _Float16* __restrict__ out,
                                int R, int C) {
    int i = blockIdx.x * blockDim.x + threadIdx.x;
    int stride = gridDim.x * blockDim.x;
    int n = R * C;
    for (; i < n; i += stride) {
        int r = i / C, c = i - r * C;
        out[c * R + r] = (_Float16)in[i];
    }
}

// ---------------- zero fill ----------------
__global__ void qigat_zero_f32(float* __restrict__ p, int n) {
    int i = blockIdx.x * blockDim.x + threadIdx.x;
    int stride = gridDim.x * blockDim.x;
    for (; i < n; i += stride) p[i] = 0.f;
}

// ---------------- WMMA GEMM: C[M,N] = A[M,K]@B[K,N] (+bias) ----------------
// A row-major f16 [M,K]; Bt = B transposed, K-major f16 [N,K]; C row-major f32.
// One wave per block computes a 16x(16*NT) C strip: 1 A-frag, NT B-frags.
// Software-pipelined over k-steps of 32. grid = (N/(16*NT), M/16).
// M%16==0, K%32==0, N%(16*NT)==0 guaranteed by caller.
template <int NT>
__global__ void qigat_wmma_gemm(const _Float16* __restrict__ A,
                                const _Float16* __restrict__ Bt,
                                const float* __restrict__ bias,
                                float* __restrict__ C,
                                int M, int K, int Nn) {
    const int l   = threadIdx.x;               // lane 0..31
    const int m0  = blockIdx.y * 16;
    const int n0  = blockIdx.x * 16 * NT;
    const int row = m0 + (l & 15);
    const int lhi8  = (l & 16) ? 8  : 0;       // A-frag K offset, upper half-wave
    const int lhi16 = (l & 16) ? 8 : 0;        // B-frag K offset, upper half-wave

    // per-lane base pointers, advanced by one k-step (32 halves = 4 v4u) per iter
    const v4u* Ap = (const v4u*)(A + (size_t)row * K + lhi8);
    const v4u* Bp[NT];
    int col[NT];
    #pragma unroll
    for (int t = 0; t < NT; ++t) {
        col[t] = n0 + 16 * t + (l & 15);
        Bp[t]  = (const v4u*)(Bt + (size_t)col[t] * K + lhi16);
    }

    v8f c[NT];
    #pragma unroll
    for (int t = 0; t < NT; ++t) c[t] = (v8f){};

    // prologue: fragments for k-step 0
    frag16 a, b[NT];
    a.u[0] = Ap[0];                            // K = lhi8 + 0..7
    a.u[1] = Ap[2];                            // K = lhi8 + 16..23
    #pragma unroll
    for (int t = 0; t < NT; ++t) {
        b[t].u[0] = Bp[t][0];                  // K = lhi16 + 0..7
        b[t].u[1] = Bp[t][2];                  // K = lhi16 + 16..23
    }

    const int steps = K >> 5;
    for (int it = 1; it < steps; ++it) {
        // preload next k-step into fresh registers (overlaps with WMMAs below)
        frag16 an, bn[NT];
        an.u[0] = Ap[4];
        an.u[1] = Ap[6];
        #pragma unroll
        for (int t = 0; t < NT; ++t) {
            bn[t].u[0] = Bp[t][4];
            bn[t].u[1] = Bp[t][6];
        }
        #pragma unroll
        for (int t = 0; t < NT; ++t)
            c[t] = __builtin_amdgcn_wmma_f32_16x16x32_f16(false, a.h, false, b[t].h,
                                                          (short)0, c[t], false, false);
        a = an;
        #pragma unroll
        for (int t = 0; t < NT; ++t) b[t] = bn[t];
        Ap += 4;
        #pragma unroll
        for (int t = 0; t < NT; ++t) Bp[t] += 4;
    }
    // epilogue k-step
    #pragma unroll
    for (int t = 0; t < NT; ++t)
        c[t] = __builtin_amdgcn_wmma_f32_16x16x32_f16(false, a.h, false, b[t].h,
                                                      (short)0, c[t], false, false);

    const int rbase = m0 + ((l & 16) ? 8 : 0);
    #pragma unroll
    for (int t = 0; t < NT; ++t) {
        const float bs = bias ? bias[col[t]] : 0.f;
        #pragma unroll
        for (int r = 0; r < 8; ++r)
            C[(size_t)(rbase + r) * Nn + col[t]] = c[t][r] + bs;
    }
}

// ---------------- column stats of z[N,128] ----------------
__global__ void qigat_colstats(const float* __restrict__ z, float* __restrict__ sum,
                               float* __restrict__ sumsq, int N) {
    int c = threadIdx.x;                   // 0..127
    float s = 0.f, sq = 0.f;
    for (int r = blockIdx.x; r < N; r += gridDim.x) {
        float v = z[r * DIN + c];
        s += v; sq += v * v;
    }
    atomicAdd(&sum[c], s);
    atomicAdd(&sumsq[c], sq);
}

// ---------------- unbiased variance + stable top-32 (rank method) ----------
__global__ void qigat_topk(const float* __restrict__ sum, const float* __restrict__ sumsq,
                           int N, int* __restrict__ topidx) {
    __shared__ float var[DIN];
    int j = threadIdx.x;
    float m = sum[j] / (float)N;
    var[j] = (sumsq[j] - sum[j] * m) / (float)(N - 1);
    __syncthreads();
    float vj = var[j];
    int rank = 0;
    for (int k = 0; k < DIN; ++k) {
        float vk = var[k];
        rank += (vk > vj) || (vk == vj && k < j);   // top_k: descending, index tiebreak
    }
    if (rank < KPAIR) topidx[rank] = j;
}

// ---------------- build Q[N,288] in f16 -----------------------------------
__global__ void qigat_qbuild(const float* __restrict__ z, const int* __restrict__ topidx,
                             _Float16* __restrict__ Q) {
    __shared__ float phi[DIN];
    __shared__ int   ti[KPAIR];
    int row = blockIdx.x;
    int t   = threadIdx.x;                 // 0..287
    if (t < KPAIR) ti[t] = topidx[t];
    if (t < DIN)   phi[t] = tanhf(z[row * DIN + t]);
    __syncthreads();
    float val;
    if (t < DIN)           val = cosf(PI_F * phi[t]);
    else if (t < 2 * DIN)  val = sinf(PI_F * phi[t - DIN]);
    else {
        int p  = t - 2 * DIN;              // pairs (0,1)..(0,31),(1,2)
        int pi = (p < 31) ? 0 : 1;
        int pj = (p < 31) ? (p + 1) : 2;
        val = cosf(phi[ti[pi]] - phi[ti[pj]]);
    }
    Q[row * TOTQ + t] = (_Float16)val;
}

// ---------------- layernorm (+elu)(+alpha)(+res) (+f16 copy) --------------
// one row per block, blockDim == L (power of two: 32/128/256)
__global__ void qigat_ln(const float* __restrict__ X, const float* __restrict__ g,
                         const float* __restrict__ b, const float* __restrict__ alpha,
                         const float* __restrict__ res, float* __restrict__ Y,
                         _Float16* __restrict__ Yh, int L, int do_elu) {
    __shared__ float red[256];
    int row = blockIdx.x, t = threadIdx.x;
    float v = X[row * L + t];
    red[t] = v; __syncthreads();
    for (int s = L >> 1; s > 0; s >>= 1) { if (t < s) red[t] += red[t + s]; __syncthreads(); }
    float mean = red[0] / (float)L; __syncthreads();
    float d = v - mean;
    red[t] = d * d; __syncthreads();
    for (int s = L >> 1; s > 0; s >>= 1) { if (t < s) red[t] += red[t + s]; __syncthreads(); }
    float var = red[0] / (float)L;
    float y = d * rsqrtf(var + LN_EPS) * g[t] + b[t];
    if (do_elu) y = elu1(y);
    if (alpha)  y *= alpha[0];
    if (res)    y += res[row * L + t];
    Y[row * L + t] = y;
    if (Yh) Yh[row * L + t] = (_Float16)y;
}

// ---------------- elementwise product (h_prod) -----------------------------
__global__ void qigat_mul(const float* __restrict__ a, const float* __restrict__ b,
                          float* __restrict__ o, int n) {
    int i = blockIdx.x * blockDim.x + threadIdx.x;
    int stride = gridDim.x * blockDim.x;
    for (; i < n; i += stride) o[i] = a[i] * b[i];
}

// ---------------- edge phase: attention + scatter-add ----------------------
// block = 128 threads = 4 waves = 4 heads (wave32); one edge per block.
__global__ void qigat_edge(const float* __restrict__ h1, const float* __restrict__ h2,
                           const float* __restrict__ hp, const int* __restrict__ src,
                           const int* __restrict__ dst, const float* __restrict__ a,
                           float* __restrict__ acc, int E) {
    int e = blockIdx.x;
    if (e >= E) return;
    int t = threadIdx.x;                   // 0..127 ; head = t>>5, chan = t&31
    int s = src[e], d0 = dst[e];
    float x1 = h1[s  * HD + t];
    float x2 = h2[d0 * HD + t];
    float p1 = hp[s  * HD + t];
    float p2 = hp[d0 * HD + t];
    float v  = elu1(x1 + x2 + p1 * p2);
    float part = a[t] * v;                 // a is [H,HID] row-major == a[t]
    #pragma unroll
    for (int off = 16; off; off >>= 1) part += __shfl_xor(part, off, 32);
    __shared__ float lg[NHEAD];
    if ((t & 31) == 0) lg[t >> 5] = part * 0.17677669529663687f;  // 1/sqrt(32)
    __syncthreads();
    float m  = fmaxf(fmaxf(lg[0], lg[1]), fmaxf(lg[2], lg[3]));
    float e0 = expf(lg[0] - m), e1 = expf(lg[1] - m);
    float e2 = expf(lg[2] - m), e3 = expf(lg[3] - m);
    float inv = 1.f / (e0 + e1 + e2 + e3);
    int   h   = t >> 5;
    float at  = (h == 0 ? e0 : h == 1 ? e1 : h == 2 ? e2 : e3) * inv;
    atomicAdd(&acc[d0 * HD + t], at * x1);
}

// ---------------- output projection 128 -> 2 -------------------------------
__global__ void qigat_out(const float* __restrict__ h, const float* __restrict__ W,
                          const float* __restrict__ b, float* __restrict__ out, int N) {
    int n = blockIdx.x * blockDim.x + threadIdx.x;
    if (n >= N) return;
    float a0 = b[0], a1 = b[1];
    #pragma unroll 4
    for (int k = 0; k < HD; ++k) {
        float v = h[n * HD + k];
        a0 += v * W[k * 2 + 0];
        a1 += v * W[k * 2 + 1];
    }
    out[n * 2 + 0] = a0;
    out[n * 2 + 1] = a1;
}

// ---------------------------------------------------------------------------
extern "C" void kernel_launch(void* const* d_in, const int* in_sizes, int n_in,
                              void* d_out, int out_size, void* d_ws, size_t ws_size,
                              hipStream_t stream) {
    // inputs in setup_inputs() order
    const float* x      = (const float*)d_in[0];
    const float* qm_W   = (const float*)d_in[1];
    const float* qm_b   = (const float*)d_in[2];
    const float* qm_cW  = (const float*)d_in[3];
    const float* qm_cb  = (const float*)d_in[4];
    const float* qm_g   = (const float*)d_in[5];
    const float* qm_be  = (const float*)d_in[6];
    const float* qm_al  = (const float*)d_in[7];
    const float* lc_W   = (const float*)d_in[8];
    const float* lc_b   = (const float*)d_in[9];
    const float* cn_g   = (const float*)d_in[10];
    const float* cn_b   = (const float*)d_in[11];
    const float* W1_0   = (const float*)d_in[12];
    const float* b1_0   = (const float*)d_in[13];
    const float* W2_0   = (const float*)d_in[14];
    const float* b2_0   = (const float*)d_in[15];
    // d_in[16],[17] = W3_0,b3_0 : computed-but-unused in reference -> skipped
    const float* a_0    = (const float*)d_in[18];
    const float* g_0    = (const float*)d_in[19];
    const float* be_0   = (const float*)d_in[20];
    const float* W1_1   = (const float*)d_in[21];
    const float* b1_1   = (const float*)d_in[22];
    const float* W2_1   = (const float*)d_in[23];
    const float* b2_1   = (const float*)d_in[24];
    // d_in[25],[26] = W3_1,b3_1 skipped
    const float* a_1    = (const float*)d_in[27];
    const float* g_1    = (const float*)d_in[28];
    const float* be_1   = (const float*)d_in[29];
    const float* out_W  = (const float*)d_in[30];
    const float* out_b  = (const float*)d_in[31];
    const int*   eidx   = (const int*)d_in[32];

    const int N = in_sizes[0] / DIN;       // 50000
    const int E = in_sizes[32] / 2;        // 800000
    const int* src = eidx;
    const int* dst = eidx + E;

    // -------- workspace layout (phase-reused) --------
    char*  ws  = (char*)d_ws;
    size_t off = 0;
    auto carve = [&](size_t bytes) -> char* {
        char* p = ws + off;
        off = (off + bytes + 255) & ~(size_t)255;
        return p;
    };
    float*    z     = (float*)   carve((size_t)N * DIN * 4);   // later: h1
    _Float16* Qh    = (_Float16*)carve((size_t)N * TOTQ * 2);  // later: h2 (N*128*4 fits)
    float*    xq    = (float*)   carve((size_t)N * QOUT * 4);  // later: hp + acc
    _Float16* xqh   = (_Float16*)carve((size_t)N * QOUT * 2);  // later: hbuf f32
    float*    xc    = (float*)   carve((size_t)N * HID * 4);
    _Float16* xch   = (_Float16*)carve((size_t)N * HID * 2);
    _Float16* hh    = (_Float16*)carve((size_t)N * HD * 2);
    _Float16* xh    = (_Float16*)carve((size_t)N * DIN * 2);
    // weights, stored TRANSPOSED (K-major) for contiguous B-fragment loads
    _Float16* qmWt  = (_Float16*)carve(DIN * DIN * 2);
    _Float16* cWt   = (_Float16*)carve(TOTQ * QOUT * 2);
    _Float16* lcWt  = (_Float16*)carve(QOUT * HID * 2);
    _Float16* W10t  = (_Float16*)carve(HID * HD * 2);
    _Float16* W20t  = (_Float16*)carve(HID * HD * 2);
    _Float16* W11t  = (_Float16*)carve(HD * HD * 2);
    _Float16* W21t  = (_Float16*)carve(HD * HD * 2);
    float*    csum  = (float*)   carve(DIN * 4);
    float*    csq   = (float*)   carve(DIN * 4);
    int*      tidx  = (int*)     carve(KPAIR * 4);
    // aliases for later phases
    float* h1   = z;                           // [N,128]
    float* h2   = (float*)Qh;                  // [N,128]
    float* hp   = xq;                          // [N,128]
    float* acc  = xq + (size_t)N * HD;         // [N,128]
    float* hbuf = (float*)xqh;                 // [N,128] layer-0 output / residual

    // -------- phase 0: f16 conversions --------
    auto cvt = [&](const float* in, _Float16* o, int n) {
        int blocks = (n + 255) / 256; if (blocks > 2048) blocks = 2048;
        qigat_cvt_f16<<<blocks, 256, 0, stream>>>(in, o, n);
    };
    auto cvtT = [&](const float* in, _Float16* o, int R, int C) {
        int n = R * C;
        int blocks = (n + 255) / 256; if (blocks > 2048) blocks = 2048;
        qigat_cvt_f16_t<<<blocks, 256, 0, stream>>>(in, o, R, C);
    };
    cvt(x, xh, N * DIN);
    cvtT(qm_W, qmWt, DIN, DIN);
    cvtT(qm_cW, cWt, TOTQ, QOUT);
    cvtT(lc_W, lcWt, QOUT, HID);
    cvtT(W1_0, W10t, HID, HD);
    cvtT(W2_0, W20t, HID, HD);
    cvtT(W1_1, W11t, HD, HD);
    cvtT(W2_1, W21t, HD, HD);

    auto gemm = [&](const _Float16* A, const _Float16* Bt, const float* bias,
                    float* C, int M, int K, int Nn) {
        if (Nn % 64 == 0) {
            dim3 grid(Nn / 64, M / 16);
            qigat_wmma_gemm<4><<<grid, 32, 0, stream>>>(A, Bt, bias, C, M, K, Nn);
        } else {
            dim3 grid(Nn / 32, M / 16);
            qigat_wmma_gemm<2><<<grid, 32, 0, stream>>>(A, Bt, bias, C, M, K, Nn);
        }
    };

    // -------- quantum mapper --------
    gemm(xh, qmWt, qm_b, z, N, DIN, DIN);                       // z = x@W + b
    qigat_zero_f32<<<1, 256, 0, stream>>>(csum, DIN);
    qigat_zero_f32<<<1, 256, 0, stream>>>(csq, DIN);
    qigat_colstats<<<512, DIN, 0, stream>>>(z, csum, csq, N);
    qigat_topk<<<1, DIN, 0, stream>>>(csum, csq, N, tidx);
    qigat_qbuild<<<N, TOTQ, 0, stream>>>(z, tidx, Qh);          // Q f16 [N,288]
    gemm(Qh, cWt, qm_cb, xq, N, TOTQ, QOUT);                    // xq = Q@cW + cb
    qigat_ln<<<N, QOUT, 0, stream>>>(xq, qm_g, qm_be, qm_al, nullptr,
                                     xq, xqh, QOUT, 0);         // LN * alpha

    // -------- compression layer --------
    gemm(xqh, lcWt, lc_b, xc, N, QOUT, HID);
    qigat_ln<<<N, HID, 0, stream>>>(xc, cn_g, cn_b, nullptr, nullptr,
                                    xc, xch, HID, 1);           // elu(LN)

    // -------- GAT layer 0 --------
    gemm(xch, W10t, b1_0, h1, N, HID, HD);
    gemm(xch, W20t, b2_0, h2, N, HID, HD);
    {
        int n = N * HD;
        qigat_mul<<<2048, 256, 0, stream>>>(h1, h2, hp, n);
        qigat_zero_f32<<<2048, 256, 0, stream>>>(acc, n);
    }
    qigat_edge<<<E, HD, 0, stream>>>(h1, h2, hp, src, dst, a_0, acc, E);
    qigat_ln<<<N, HD, 0, stream>>>(acc, g_0, be_0, nullptr, nullptr,
                                   hbuf, hh, HD, 1);            // elu(LN) -> residual

    // -------- GAT layer 1 --------
    gemm(hh, W11t, b1_1, h1, N, HD, HD);
    gemm(hh, W21t, b2_1, h2, N, HD, HD);
    {
        int n = N * HD;
        qigat_mul<<<2048, 256, 0, stream>>>(h1, h2, hp, n);
        qigat_zero_f32<<<2048, 256, 0, stream>>>(acc, n);
    }
    qigat_edge<<<E, HD, 0, stream>>>(h1, h2, hp, src, dst, a_1, acc, E);
    qigat_ln<<<N, HD, 0, stream>>>(acc, g_1, be_1, nullptr, hbuf,
                                   acc, nullptr, HD, 1);        // elu(LN)+res, in place

    // -------- output projection --------
    qigat_out<<<(N + 255) / 256, 256, 0, stream>>>(acc, out_W, out_b,
                                                   (float*)d_out, N);
    (void)n_in; (void)out_size; (void)ws_size;
}
